// CausalSelfAttentionHead_50500225466982
// MI455X (gfx1250) — matrix operand order repaired and checked by reading, not compiled, not run on hardware
//
#include <hip/hip_runtime.h>

// CDNA5 / gfx1250 causal self-attention (single head), bf16 WMMA + flash softmax.
// Kernel 1: QKV projection GEMM (bf16 operands, fp32 accumulate) -> bf16 q/k/v in d_ws.
// Kernel 2: flash attention, online softmax, fp32 output.
// All WMMA fragments are fetched from LDS as aligned 16B vector loads (ds_load_b128).

typedef __attribute__((ext_vector_type(16))) __bf16 v16bf;
typedef __attribute__((ext_vector_type(8)))  __bf16 v8bf;
typedef __attribute__((ext_vector_type(4)))  __bf16 v4bf;
typedef __attribute__((ext_vector_type(8)))  float  v8f;

constexpr int B = 4, T = 2048, C = 1024, H = 64;
constexpr int BT = B * T;

// ---------------------------------------------------------------------------
// Fragment loads from LDS (wave32), in this file's own K order, not the ISA table's: all K
// runs contiguous so each fragment is two aligned 16-byte LDS loads.
// A (16x32 bf16): lane: row m=lane%16, half hh=lane/16;
//   VGPR j=0..7 : K = kbase + hh*16 + (0..15)      (16 contiguous bf16)
// B (32x16 bf16), tile stored n-major (tile[n][k]): lane: col n=lane%16;
//   VGPR j=0..7 : K = kbase + hh*16 + (0..15)      (16 contiguous bf16)
// C/D (16x16 f32): VGPR j -> row j / j+8 by lane half, col = lane%16
// ---------------------------------------------------------------------------

__device__ inline v16bf ld_a_frag(const __bf16* tile, int lda, int row0, int kbase) {
  const int lane = threadIdx.x & 31;
  const int m = lane & 15, hh = lane >> 4;
  const __bf16* p = tile + (row0 + m) * lda + kbase + hh * 16;
  const v8bf lo = *(const v8bf*)(p);        // VGPR 0..3
  const v8bf hi = *(const v8bf*)(p + 8);    // VGPR 4..7
  return __builtin_shufflevector(lo, hi, 0, 1, 2, 3, 4, 5, 6, 7,
                                         8, 9, 10, 11, 12, 13, 14, 15);
}

// B-fragment from an n-major tile: B[k][n] = tile[n][k].
__device__ inline v16bf ld_b_frag(const __bf16* tile, int ld, int n0, int kbase) {
  const int lane = threadIdx.x & 31;
  const int n = lane & 15, hh = lane >> 4;
  const __bf16* p = tile + (n0 + n) * ld + kbase + hh * 16;
  const v8bf lo = *(const v8bf*)(p);
  const v8bf hi = *(const v8bf*)(p + 8);
  return __builtin_shufflevector(lo, hi, 0, 1, 2, 3, 4, 5, 6, 7,
                                         8, 9, 10, 11, 12, 13, 14, 15);
}

// ---------------------------------------------------------------------------
// Kernel 1: QKV projection. grid=(BT/64, 3), block=128 (4 waves).
// Block computes a 64x64 output tile of q (y=0), k (y=1) or v (y=2).
// ---------------------------------------------------------------------------
__global__ __launch_bounds__(128) void qkv_proj_kernel(
    const float* __restrict__ X,
    const float* __restrict__ Wq, const float* __restrict__ bq,
    const float* __restrict__ Wk, const float* __restrict__ bk,
    const float* __restrict__ Wv, const float* __restrict__ bv,
    __bf16* __restrict__ qo, __bf16* __restrict__ ko, __bf16* __restrict__ vo) {
  __shared__ alignas(16) __bf16 sA[64][40];    // 64 rows x 32 K-chunk (+pad), 80B rows
  __shared__ alignas(16) __bf16 sBt[64][40];   // W transposed: sBt[n][k]

  const int tid  = threadIdx.x;
  const int lane = tid & 31;
  const int w    = tid >> 5;
  const int row0 = blockIdx.x * 64;

  const float* Wm;
  const float* bias;
  __bf16* out;
  if (blockIdx.y == 0)      { Wm = Wq; bias = bq; out = qo; }
  else if (blockIdx.y == 1) { Wm = Wk; bias = bk; out = ko; }
  else                      { Wm = Wv; bias = bv; out = vo; }

  const v8f vzero = {0.f, 0.f, 0.f, 0.f, 0.f, 0.f, 0.f, 0.f};
  v8f acc[4];
#pragma unroll
  for (int c = 0; c < 4; ++c) acc[c] = vzero;

  for (int kb = 0; kb < C / 32; ++kb) {
    // input chunk 64x32 fp32 -> bf16, vectorized float4 loads (512 per chunk)
#pragma unroll
    for (int t = 0; t < 4; ++t) {
      const int vi = tid + t * 128;
      const int r = vi >> 3, c4 = vi & 7;                   // 8 float4 per row
      const float4 v = *(const float4*)&X[(size_t)(row0 + r) * C + kb * 32 + c4 * 4];
      v4bf pk = {(__bf16)v.x, (__bf16)v.y, (__bf16)v.z, (__bf16)v.w};
      *(v4bf*)&sA[r][c4 * 4] = pk;
    }
    // weight chunk 32x64 fp32 -> bf16, stored transposed (n-major)
#pragma unroll
    for (int t = 0; t < 4; ++t) {
      const int vi = tid + t * 128;
      const int r = vi >> 4, c4 = vi & 15;                  // 16 float4 per row
      const float4 v = *(const float4*)&Wm[(size_t)(kb * 32 + r) * H + c4 * 4];
      sBt[c4 * 4 + 0][r] = (__bf16)v.x;
      sBt[c4 * 4 + 1][r] = (__bf16)v.y;
      sBt[c4 * 4 + 2][r] = (__bf16)v.z;
      sBt[c4 * 4 + 3][r] = (__bf16)v.w;
    }
    // prefetch next input chunk
    if (kb + 1 < C / 32)
      __builtin_prefetch(&X[(size_t)(row0 + (tid >> 1)) * C + (kb + 1) * 32 + (tid & 1) * 16], 0, 1);
    __syncthreads();

    const v16bf a = ld_a_frag(&sA[0][0], 40, w * 16, 0);
#pragma unroll
    for (int c = 0; c < 4; ++c) {
      const v16bf b = ld_b_frag(&sBt[0][0], 40, c * 16, 0);
      acc[c] = __builtin_amdgcn_wmma_f32_16x16x32_bf16(
          false, a, false, b, (short)0, acc[c], false, false);
    }
    __syncthreads();
  }

  // epilogue: bias add, cast, store bf16
#pragma unroll
  for (int c = 0; c < 4; ++c) {
#pragma unroll
    for (int j = 0; j < 8; ++j) {
      const int rl  = (lane < 16) ? j : j + 8;
      const int col = c * 16 + (lane & 15);
      const float v = acc[c][j] + bias[col];
      out[(size_t)(row0 + w * 16 + rl) * H + col] = (__bf16)v;
    }
  }
}

// ---------------------------------------------------------------------------
// Kernel 2: flash attention. grid=(T/64, B), block=128 (4 waves).
// Block owns 64 query rows of one batch; wave w owns rows [w*16, w*16+16).
// ---------------------------------------------------------------------------
__global__ __launch_bounds__(128) void flash_attn_kernel(
    const __bf16* __restrict__ Qg, const __bf16* __restrict__ Kg,
    const __bf16* __restrict__ Vg, float* __restrict__ Og) {
  __shared__ alignas(16) __bf16 sQ[64][72];        // row-major [q][h]
  __shared__ alignas(16) __bf16 sK[64][72];        // row-major [s][h]  (n-major for S B-frag)
  __shared__ alignas(16) __bf16 sVt[64][72];       // transposed [h][s] (n-major for PV B-frag)
  __shared__ alignas(16) float  sS[4][16][68];     // per-wave S slab, 272B rows
  __shared__ alignas(16) __bf16 sP[4][16][72];     // per-wave P slab (A-matrix of P·V)
  __shared__ float sRow[64];                       // per-row correction / 1/l broadcast

  const int tid  = threadIdx.x;
  const int lane = tid & 31;
  const int w    = tid >> 5;
  const int qblk = blockIdx.x;
  const int bb   = blockIdx.y;
  const int q0   = qblk * 64;
  const size_t base = (size_t)bb * T * H;
  const float scale = 0.125f;                      // 1/sqrt(H), H=64
  const float NEG_INF = -__builtin_inff();

  // load Q tile once (b128 loads: 8 bf16 per access)
#pragma unroll
  for (int t = 0; t < 4; ++t) {
    const int vi = tid + t * 128;
    const int r = vi >> 3, c8 = vi & 7;
    *(v8bf*)&sQ[r][c8 * 8] = *(const v8bf*)&Qg[base + (size_t)(q0 + r) * H + c8 * 8];
  }
  __syncthreads();

  // Q fragments are loop-invariant: hoist out of the key loop
  const v16bf aq0 = ld_a_frag(&sQ[0][0], 72, w * 16, 0);
  const v16bf aq1 = ld_a_frag(&sQ[0][0], 72, w * 16, 32);

  const v8f vzero = {0.f, 0.f, 0.f, 0.f, 0.f, 0.f, 0.f, 0.f};
  v8f O[4];
#pragma unroll
  for (int c = 0; c < 4; ++c) O[c] = vzero;
  float m_i = NEG_INF;   // running max (valid in lanes 0-15, row = lane)
  float l_i = 0.f;       // running sum

  for (int j = 0; j <= qblk; ++j) {
    const int k0 = j * 64;
    // K row-major; V transposed into sVt[h][s]
#pragma unroll
    for (int t = 0; t < 4; ++t) {
      const int vi = tid + t * 128;
      const int r = vi >> 3, c8 = vi & 7;
      *(v8bf*)&sK[r][c8 * 8] = *(const v8bf*)&Kg[base + (size_t)(k0 + r) * H + c8 * 8];
      const v8bf vv = *(const v8bf*)&Vg[base + (size_t)(k0 + r) * H + c8 * 8];
#pragma unroll
      for (int e = 0; e < 8; ++e) sVt[c8 * 8 + e][r] = vv[e];
    }
    // prefetch next K/V block while this one is consumed
    if (j < qblk) {
      const int pr = tid >> 1, pc = (tid & 1) * 32;
      __builtin_prefetch(&Kg[base + (size_t)(k0 + 64 + pr) * H + pc], 0, 1);
      __builtin_prefetch(&Vg[base + (size_t)(k0 + 64 + pr) * H + pc], 0, 1);
    }
    __syncthreads();

    // S = Q · K^T for this wave's 16 rows (4 col-tiles, Hdim=64 -> 2 WMMAs each)
#pragma unroll
    for (int c = 0; c < 4; ++c) {
      v8f s = vzero;
      const v16bf b0 = ld_b_frag(&sK[0][0], 72, c * 16, 0);
      s = __builtin_amdgcn_wmma_f32_16x16x32_bf16(false, aq0, false, b0, (short)0, s, false, false);
      const v16bf b1 = ld_b_frag(&sK[0][0], 72, c * 16, 32);
      s = __builtin_amdgcn_wmma_f32_16x16x32_bf16(false, aq1, false, b1, (short)0, s, false, false);
      // scale + causal mask, spill S slab to LDS (wave-private)
#pragma unroll
      for (int jj = 0; jj < 8; ++jj) {
        const int rl  = (lane < 16) ? jj : jj + 8;
        const int col = c * 16 + (lane & 15);
        float v = s[jj] * scale;
        if (k0 + col > q0 + w * 16 + rl) v = NEG_INF;
        sS[w][rl][col] = v;
      }
    }
    __builtin_amdgcn_wave_barrier();   // pin DS write->read ordering (in-order per wave)

    // online softmax: lanes 0-15 each own one row of the wave's 16-row slab
    if (lane < 16) {
      const float4* srow = (const float4*)&sS[w][lane][0];
      float rmax = NEG_INF;
#pragma unroll
      for (int n4 = 0; n4 < 16; ++n4) {
        const float4 v = srow[n4];
        rmax = fmaxf(rmax, fmaxf(fmaxf(v.x, v.y), fmaxf(v.z, v.w)));
      }
      const float m_new = fmaxf(m_i, rmax);
      const float corr  = __expf(m_i - m_new);
      float rsum = 0.f;
#pragma unroll
      for (int n4 = 0; n4 < 16; ++n4) {
        const float4 v = srow[n4];
        const float p0 = __expf(v.x - m_new);
        const float p1 = __expf(v.y - m_new);
        const float p2 = __expf(v.z - m_new);
        const float p3 = __expf(v.w - m_new);
        v4bf pk = {(__bf16)p0, (__bf16)p1, (__bf16)p2, (__bf16)p3};
        *(v4bf*)&sP[w][lane][n4 * 4] = pk;
        rsum += (p0 + p1) + (p2 + p3);
      }
      l_i = l_i * corr + rsum;
      m_i = m_new;
      sRow[w * 16 + lane] = corr;
    }
    __builtin_amdgcn_wave_barrier();

    // rescale O accumulators by per-row correction
#pragma unroll
    for (int c = 0; c < 4; ++c) {
#pragma unroll
      for (int jj = 0; jj < 8; ++jj) {
        const int rl = (lane < 16) ? jj : jj + 8;
        O[c][jj] *= sRow[w * 16 + rl];
      }
    }

    // O += P · V  (K-dim = 64 keys -> 2 WMMAs per 16x16 tile)
    const v16bf ap0 = ld_a_frag(&sP[w][0][0], 72, 0, 0);
    const v16bf ap1 = ld_a_frag(&sP[w][0][0], 72, 0, 32);
#pragma unroll
    for (int c = 0; c < 4; ++c) {
      const v16bf bv0 = ld_b_frag(&sVt[0][0], 72, c * 16, 0);
      O[c] = __builtin_amdgcn_wmma_f32_16x16x32_bf16(false, ap0, false, bv0, (short)0, O[c], false, false);
      const v16bf bv1 = ld_b_frag(&sVt[0][0], 72, c * 16, 32);
      O[c] = __builtin_amdgcn_wmma_f32_16x16x32_bf16(false, ap1, false, bv1, (short)0, O[c], false, false);
    }
    __syncthreads();
  }

  // final 1/l normalization and fp32 store
  if (lane < 16) sRow[w * 16 + lane] = 1.f / l_i;
  __builtin_amdgcn_wave_barrier();
#pragma unroll
  for (int c = 0; c < 4; ++c) {
#pragma unroll
    for (int jj = 0; jj < 8; ++jj) {
      const int rl = (lane < 16) ? jj : jj + 8;
      const float invl = sRow[w * 16 + rl];
      Og[base + (size_t)(q0 + w * 16 + rl) * H + c * 16 + (lane & 15)] = O[c][jj] * invl;
    }
  }
}

// ---------------------------------------------------------------------------
// Host launcher
// ---------------------------------------------------------------------------
extern "C" void kernel_launch(void* const* d_in, const int* in_sizes, int n_in,
                              void* d_out, int out_size, void* d_ws, size_t ws_size,
                              hipStream_t stream) {
  const float* X  = (const float*)d_in[0];
  const float* Wq = (const float*)d_in[1];
  const float* bq = (const float*)d_in[2];
  const float* Wk = (const float*)d_in[3];
  const float* bk = (const float*)d_in[4];
  const float* Wv = (const float*)d_in[5];
  const float* bv = (const float*)d_in[6];

  __bf16* qb = (__bf16*)d_ws;          // [BT, H] bf16
  __bf16* kb = qb + (size_t)BT * H;    // [BT, H] bf16
  __bf16* vb = kb + (size_t)BT * H;    // [BT, H] bf16  (total 3 MB of ws)

  dim3 g1(BT / 64, 3);
  qkv_proj_kernel<<<g1, 128, 0, stream>>>(X, Wq, bq, Wk, bk, Wv, bv, qb, kb, vb);

  dim3 g2(T / 64, B);
  flash_attn_kernel<<<g2, 128, 0, stream>>>(qb, kb, vb, (float*)d_out);
}
